// MAA_38568806318777
// MI455X (gfx1250) — matrix operand order repaired and checked by reading, not compiled, not run on hardware
//
#include <hip/hip_runtime.h>
#include <hip/hip_bf16.h>
#include <math.h>

// ---------------------------------------------------------------------------
// Memory-augmented attention for MI455X (gfx1250), bf16 WMMA path.
// b=2, n=2048, m=256, DIM=1024, HEADS=16, DHEAD=64, L=n+m=2304
// GEMMs: 16x64 tile per wave; per 32-wide k-step all 5 fragments (1 A + 4 B)
// are loaded before the 4 WMMAs so loads overlap matrix issue via partial
// s_wait_loadcnt.  Single-buffered fragments (no spills).
// Attention: flash-style, S^T = K*Q^T, P staged via LDS, V^T contiguous.
// ---------------------------------------------------------------------------

typedef __attribute__((ext_vector_type(16))) __bf16 v16bf;
typedef __attribute__((ext_vector_type(8)))  float  v8f;

#define DIMC   1024
#define NQ     2048
#define NM     256
#define LKV    2304   // 2048 + 256
#define NHEAD  16
#define DHEAD  64
#define NB     2

union AFrag { v16bf v; __bf16 e[16]; unsigned u[8]; };

// A fragment (16x32 bf16): lane holds row M=lane&15.
// VGPR v: K = (v<4 ? 0 : 16) + (lane>=16 ? 8 : 0) + (v&3)*2, pair {K,K+1}.
__device__ inline v16bf load_A_rowmajor(const __bf16* __restrict src, int ld,
                                        int k0, int lane) {
  AFrag a;
  const int M = lane & 15, half = lane >> 4;
  const __bf16* row = src + (size_t)M * ld + k0;
#pragma unroll
  for (int v = 0; v < 8; ++v) {
    const int K = ((v < 4) ? 0 : 16) + half * 8 + (v & 3) * 2;
    a.u[v] = *(const unsigned*)(row + K);
  }
  return a.v;
}

// B fragment (32x16 bf16) from a TRANSPOSED source (row = n, contiguous in k):
// B[k][n] = srcT[(n0+n)*ld + k0 + k]  -> 8 contiguous dword loads per lane.
// lane holds col N=lane&15; VGPR v: K = (v<4 ? 0 : 16) + (lane>=16 ? 8 : 0) + (v&3)*2.
__device__ inline v16bf load_B_transposed(const __bf16* __restrict srcT, int ld,
                                          int k0, int n0, int lane) {
  AFrag b;
  const int N = lane & 15, half = lane >> 4;
  const __bf16* row = srcT + (size_t)(n0 + N) * ld + k0;
#pragma unroll
  for (int v = 0; v < 8; ++v)
    b.u[v] = *(const unsigned*)(row + ((v < 4) ? 0 : 16) + half * 8 + (v & 3) * 2);
  return b.v;
}

// ---------------------------------------------------------------------------
// fp32 (K x N) -> bf16 transposed (N x K).  Coalesced reads along n.
// ---------------------------------------------------------------------------
__global__ void cvt_transpose_bf16(const float* __restrict src,
                                   __bf16* __restrict dstT, int K, int N) {
  const int n = blockIdx.x * 256 + threadIdx.x;
  const int k = blockIdx.y;
  if (n < N) dstT[(size_t)n * K + k] = (__bf16)src[(size_t)k * N + n];
}

// ---------------------------------------------------------------------------
// fp32 -> bf16 (row-major copy, for memories)
// ---------------------------------------------------------------------------
__global__ void cvt_f32_bf16(const float* __restrict src, __bf16* __restrict dst,
                             int n) {
  int i = blockIdx.x * 256 + threadIdx.x;
  if (i < n) dst[i] = (__bf16)src[i];
}

// ---------------------------------------------------------------------------
// LayerNorm(1024) -> bf16 normalized activations (row-major)
// ---------------------------------------------------------------------------
__global__ void layernorm_bf16(const float* __restrict x, const float* __restrict g,
                               const float* __restrict bta, __bf16* __restrict out) {
  const int row = blockIdx.x;
  const float* xr = x + (size_t)row * DIMC;
  __shared__ float red[256];

  float s = 0.f;
  for (int i = threadIdx.x; i < DIMC; i += 256) s += xr[i];
  red[threadIdx.x] = s;
  __syncthreads();
  for (int st = 128; st > 0; st >>= 1) {
    if (threadIdx.x < st) red[threadIdx.x] += red[threadIdx.x + st];
    __syncthreads();
  }
  const float mu = red[0] * (1.0f / DIMC);
  __syncthreads();

  float v = 0.f;
  for (int i = threadIdx.x; i < DIMC; i += 256) {
    float t = xr[i] - mu;
    v += t * t;
  }
  red[threadIdx.x] = v;
  __syncthreads();
  for (int st = 128; st > 0; st >>= 1) {
    if (threadIdx.x < st) red[threadIdx.x] += red[threadIdx.x + st];
    __syncthreads();
  }
  const float rstd = rsqrtf(red[0] * (1.0f / DIMC) + 1e-5f);

  __bf16* orow = out + (size_t)row * DIMC;
  for (int i = threadIdx.x; i < DIMC; i += 256)
    orow[i] = (__bf16)((xr[i] - mu) * rstd * g[i] + bta[i]);
}

// ---------------------------------------------------------------------------
// Projection GEMM: C = A(bf16, rows x K) * W (via W^T, bf16 N x K).
// One wave computes a 16x64 tile.  Epilogue scatters into head-split layouts:
//   transposedDst==0: dst[((b*16+h)*Ldst + jOff + q)*64 + d]       (Q, K)
//   transposedDst==1: dst[(((b*16+h)*64 + d))*Ldst + jOff + q]     (V^T)
// ---------------------------------------------------------------------------
__global__ void proj_gemm_bf16(const __bf16* __restrict A, const __bf16* __restrict WT,
                               __bf16* __restrict dst, int K, int colOff,
                               int rowsShift, int Ldst, int jOff, int transposedDst) {
  const int lane = threadIdx.x;
  const int row0 = blockIdx.x * 16;
  const int col0 = blockIdx.y * 64;

  const __bf16* Abase = A + (size_t)row0 * K;
  const int n0 = colOff + col0;

  v8f c[4] = {{}, {}, {}, {}};

#pragma unroll 2
  for (int k0 = 0; k0 < K; k0 += 32) {
    // Load all fragments for this k-step up front so the 4 WMMAs can start
    // on partial s_wait_loadcnt while later B fragments are still in flight.
    v16bf a = load_A_rowmajor(Abase, K, k0, lane);
    v16bf b[4];
#pragma unroll
    for (int j = 0; j < 4; ++j)
      b[j] = load_B_transposed(WT, K, k0, n0 + j * 16, lane);
    if (k0 + 32 < K) {
      __builtin_prefetch(Abase + (size_t)(lane & 15) * K + k0 + 32, 0, 3);
      __builtin_prefetch(WT + (size_t)(n0 + (lane & 15)) * K + k0 + 32, 0, 3);
    }
#pragma unroll
    for (int j = 0; j < 4; ++j)
      c[j] = __builtin_amdgcn_wmma_f32_16x16x32_bf16(false, a, false, b[j], (short)0,
                                                     c[j], false, false);
  }

  const int N = lane & 15, half = lane >> 4;
  const int rowsMask = (1 << rowsShift) - 1;
#pragma unroll
  for (int j = 0; j < 4; ++j) {
    const int gN = col0 + j * 16 + N;
    const int h = gN >> 6, d = gN & 63;
#pragma unroll
    for (int r = 0; r < 8; ++r) {
      const int gM = row0 + half * 8 + r;
      const int bb = gM >> rowsShift;
      const int q  = gM & rowsMask;
      size_t idx;
      if (transposedDst)
        idx = (((size_t)(bb * NHEAD + h)) * DHEAD + d) * Ldst + jOff + q;
      else
        idx = (((size_t)(bb * NHEAD + h)) * Ldst + jOff + q) * DHEAD + d;
      dst[idx] = (__bf16)c[j][r];
    }
  }
}

// ---------------------------------------------------------------------------
// Output GEMM: out(f32) = A(bf16, rows x 1024) * Wo (via Wo^T) + bo.
// Same 16x64 tiling.
// ---------------------------------------------------------------------------
__global__ void out_gemm_f32(const __bf16* __restrict A, const __bf16* __restrict WoT,
                             const float* __restrict bias, float* __restrict out) {
  const int lane = threadIdx.x;
  const int row0 = blockIdx.x * 16;
  const int col0 = blockIdx.y * 64;

  const __bf16* Abase = A + (size_t)row0 * DIMC;

  v8f c[4] = {{}, {}, {}, {}};

#pragma unroll 2
  for (int k0 = 0; k0 < DIMC; k0 += 32) {
    v16bf a = load_A_rowmajor(Abase, DIMC, k0, lane);
    v16bf b[4];
#pragma unroll
    for (int j = 0; j < 4; ++j)
      b[j] = load_B_transposed(WoT, DIMC, k0, col0 + j * 16, lane);
    if (k0 + 32 < DIMC) {
      __builtin_prefetch(Abase + (size_t)(lane & 15) * DIMC + k0 + 32, 0, 3);
      __builtin_prefetch(WoT + (size_t)(col0 + (lane & 15)) * DIMC + k0 + 32, 0, 3);
    }
#pragma unroll
    for (int j = 0; j < 4; ++j)
      c[j] = __builtin_amdgcn_wmma_f32_16x16x32_bf16(false, a, false, b[j], (short)0,
                                                     c[j], false, false);
  }

  const int N = lane & 15, half = lane >> 4;
#pragma unroll
  for (int j = 0; j < 4; ++j) {
    const int gN = col0 + j * 16 + N;
    const float bv = bias[gN];
#pragma unroll
    for (int r = 0; r < 8; ++r) {
      const int gM = row0 + half * 8 + r;
      out[(size_t)gM * DIMC + gN] = c[j][r] + bv;
    }
  }
}

// ---------------------------------------------------------------------------
// Flash-style attention. One wave handles 16 queries for one (b,h).
// S^T = K*Q^T so per-query softmax stats are in-lane (+ one shfl_xor 16).
// P is staged through LDS to A-fragment layout; V is read transposed
// (V^T[d][j], contiguous in j) so its B fragments are dword loads.
// ---------------------------------------------------------------------------
__global__ void attn_kernel(const __bf16* __restrict Qb, const __bf16* __restrict Kb,
                            const __bf16* __restrict VTb, __bf16* __restrict Ob) {
  const int lane = threadIdx.x;
  const int q0 = blockIdx.x * 16;
  const int h = blockIdx.y, b = blockIdx.z;
  const int bh = b * NHEAD + h;

  const __bf16* Qh  = Qb  + (size_t)bh * NQ * DHEAD;
  const __bf16* Kh  = Kb  + (size_t)bh * LKV * DHEAD;
  const __bf16* VTh = VTb + (size_t)bh * DHEAD * LKV;   // [d][j]

  __shared__ __bf16 sP[16 * 32];
  __shared__ float sA[16];
  __shared__ float sL[16];

  const float scale = 0.125f;  // 1/sqrt(64)
  const int half = lane >> 4, qn = lane & 15;

  // Q fragments are loop-invariant: load once.
  v16bf bq0 = load_B_transposed(Qh + (size_t)q0 * DHEAD, DHEAD, 0, 0, lane);
  v16bf bq1 = load_B_transposed(Qh + (size_t)q0 * DHEAD, DHEAD, 32, 0, lane);

  v8f O[4] = {{}, {}, {}, {}};
  float m = -1e30f, l = 0.f;

  for (int j0 = 0; j0 < LKV; j0 += 32) {
    // --- two 16x16 S^T tiles: rows = keys, cols = queries -----------------
    v8f st[2];
#pragma unroll
    for (int t = 0; t < 2; ++t) {
      const __bf16* Kt = Kh + (size_t)(j0 + t * 16) * DHEAD;
      v8f c = {};
      v16bf a0 = load_A_rowmajor(Kt, DHEAD, 0, lane);
      v16bf a1 = load_A_rowmajor(Kt, DHEAD, 32, lane);
      c = __builtin_amdgcn_wmma_f32_16x16x32_bf16(false, a0, false, bq0, (short)0,
                                                  c, false, false);
      c = __builtin_amdgcn_wmma_f32_16x16x32_bf16(false, a1, false, bq1, (short)0,
                                                  c, false, false);
      st[t] = c;
    }

    // --- online softmax stats for query qn (this lane's column) ----------
    float tmax = -1e30f;
#pragma unroll
    for (int t = 0; t < 2; ++t)
#pragma unroll
      for (int r = 0; r < 8; ++r) {
        st[t][r] *= scale;
        tmax = fmaxf(tmax, st[t][r]);
      }
    tmax = fmaxf(tmax, __shfl_xor(tmax, 16, 32));
    const float mnew = fmaxf(m, tmax);
    const float alpha = __expf(m - mnew);
    float tsum = 0.f;
#pragma unroll
    for (int t = 0; t < 2; ++t)
#pragma unroll
      for (int r = 0; r < 8; ++r) {
        const float p = __expf(st[t][r] - mnew);
        st[t][r] = p;
        tsum += p;
      }
    tsum += __shfl_xor(tsum, 16, 32);
    l = l * alpha + tsum;
    m = mnew;

    if (half == 0) sA[qn] = alpha;
    __syncthreads();

    // --- rescale O accumulators (query of VGPR r is half*8+r) -------------
    float ar[8];
#pragma unroll
    for (int r = 0; r < 8; ++r) ar[r] = sA[half * 8 + r];
#pragma unroll
    for (int dt = 0; dt < 4; ++dt)
#pragma unroll
      for (int r = 0; r < 8; ++r) O[dt][r] *= ar[r];

    // --- stage P tile to LDS in [query][32 keys] layout -------------------
#pragma unroll
    for (int t = 0; t < 2; ++t)
#pragma unroll
      for (int r = 0; r < 8; r += 2) {
        union { __bf16 hv[2]; unsigned u; } pk;
        pk.hv[0] = (__bf16)st[t][r];
        pk.hv[1] = (__bf16)st[t][r + 1];
        *(unsigned*)&sP[qn * 32 + t * 16 + half * 8 + r] = pk.u;
      }
    __syncthreads();

    // --- O += P(16x32) * V(32x64), 4 column tiles of 16 -------------------
    v16bf ap = load_A_rowmajor(sP, 32, 0, lane);
#pragma unroll
    for (int dt = 0; dt < 4; ++dt) {
      v16bf bv = load_B_transposed(VTh, LKV, j0, dt * 16, lane);
      O[dt] = __builtin_amdgcn_wmma_f32_16x16x32_bf16(false, ap, false, bv,
                                                      (short)0, O[dt], false, false);
    }
    __syncthreads();  // protect sP before next chunk overwrites it
  }

  // --- normalize by l and store bf16 into (b, q, h*64+d) layout -----------
  if (half == 0) sL[qn] = l;
  __syncthreads();
  float li[8];
#pragma unroll
  for (int r = 0; r < 8; ++r) li[r] = 1.0f / sL[half * 8 + r];

#pragma unroll
  for (int dt = 0; dt < 4; ++dt)
#pragma unroll
    for (int r = 0; r < 8; ++r) {
      const int qg = q0 + half * 8 + r;
      const int d = dt * 16 + qn;
      const size_t idx = ((size_t)(b * NQ + qg)) * DIMC + h * DHEAD + d;
      Ob[idx] = (__bf16)(O[dt][r] * li[r]);
    }
}

// ---------------------------------------------------------------------------
// Host launcher
// ---------------------------------------------------------------------------
extern "C" void kernel_launch(void* const* d_in, const int* in_sizes, int n_in,
                              void* d_out, int out_size, void* d_ws, size_t ws_size,
                              hipStream_t stream) {
  (void)in_sizes; (void)n_in; (void)out_size; (void)ws_size;

  const float* x    = (const float*)d_in[0];
  const float* mem  = (const float*)d_in[1];
  const float* ln_g = (const float*)d_in[2];
  const float* ln_b = (const float*)d_in[3];
  const float* Wq   = (const float*)d_in[4];
  const float* Wkv  = (const float*)d_in[5];
  const float* Wm   = (const float*)d_in[6];
  const float* Wo   = (const float*)d_in[7];
  const float* bo   = (const float*)d_in[8];
  float* out = (float*)d_out;

  // workspace carve-up (bytes)
  char* ws = (char*)d_ws;
  __bf16* xn_bf  = (__bf16*)(ws);                       // 4096x1024        8 MB
  __bf16* wqT    = (__bf16*)(ws + 8388608);             // 1024x1024 (T)    2 MB
  __bf16* wkvT   = (__bf16*)(ws + 10485760);            // 2048x1024 (T)    4 MB
  __bf16* wmT    = (__bf16*)(ws + 14680064);            // 1024x1024 (T)    2 MB
  __bf16* woT    = (__bf16*)(ws + 16777216);            // 1024x1024 (T)    2 MB
  __bf16* mem_bf = (__bf16*)(ws + 18874368);            // 512x1024         1 MB
  __bf16* q_bf   = (__bf16*)(ws + 19922944);            // (2,16,2048,64)   8 MB
  __bf16* k_bf   = (__bf16*)(ws + 28311552);            // (2,16,2304,64)   9 MB
  __bf16* vT_bf  = (__bf16*)(ws + 37748736);            // (2,16,64,2304)   9 MB
  __bf16* at_bf  = (__bf16*)(ws + 47185920);            // 4096x1024        8 MB

  const int ROWS = NB * NQ;  // 4096

  // 1) weight conversion + transpose (W^T[n][k], contiguous in k)
  cvt_transpose_bf16<<<dim3(DIMC / 256, DIMC), 256, 0, stream>>>(Wq, wqT, DIMC, DIMC);
  cvt_transpose_bf16<<<dim3(2 * DIMC / 256, DIMC), 256, 0, stream>>>(Wkv, wkvT, DIMC,
                                                                     2 * DIMC);
  cvt_transpose_bf16<<<dim3(DIMC / 256, DIMC), 256, 0, stream>>>(Wm, wmT, DIMC, DIMC);
  cvt_transpose_bf16<<<dim3(DIMC / 256, DIMC), 256, 0, stream>>>(Wo, woT, DIMC, DIMC);
  cvt_f32_bf16<<<(524288 + 255) / 256, 256, 0, stream>>>(mem, mem_bf, 524288);

  // 2) LayerNorm -> bf16
  layernorm_bf16<<<ROWS, 256, 0, stream>>>(x, ln_g, ln_b, xn_bf);

  // 3) projections (WMMA).  16x64 tiles: grid.y = 1024/64 = 16.
  proj_gemm_bf16<<<dim3(ROWS / 16, DIMC / 64), 32, 0, stream>>>(
      xn_bf, wqT, q_bf, DIMC, 0, 11, NQ, 0, 0);
  proj_gemm_bf16<<<dim3(ROWS / 16, DIMC / 64), 32, 0, stream>>>(
      xn_bf, wkvT, k_bf, DIMC, 0, 11, LKV, 0, 0);
  proj_gemm_bf16<<<dim3(ROWS / 16, DIMC / 64), 32, 0, stream>>>(
      xn_bf, wkvT, vT_bf, DIMC, DIMC, 11, LKV, 0, 1);
  // memories -> appended K and V^T slots (j >= 2048)
  proj_gemm_bf16<<<dim3((NB * NM) / 16, DIMC / 64), 32, 0, stream>>>(
      mem_bf, wmT, k_bf, DIMC, 0, 8, LKV, NQ, 0);
  proj_gemm_bf16<<<dim3((NB * NM) / 16, DIMC / 64), 32, 0, stream>>>(
      mem_bf, wmT, vT_bf, DIMC, 0, 8, LKV, NQ, 1);

  // 4) attention (WMMA flash-style)
  attn_kernel<<<dim3(NQ / 16, NHEAD, NB), 32, 0, stream>>>(q_bf, k_bf, vT_bf, at_bf);

  // 5) output projection + bias (WMMA, fp32 out)
  out_gemm_f32<<<dim3(ROWS / 16, DIMC / 64), 32, 0, stream>>>(at_bf, woT, bo, out);
}
